// DurationPredictorGRU_58523224375975
// MI455X (gfx1250) — matrix operand order repaired and checked by reading, not compiled, not run on hardware
//
#include <hip/hip_runtime.h>
#include <hip/hip_bf16.h>

// ---------------------------------------------------------------------------
// Bidirectional GRU duration predictor for MI455X (gfx1250, wave32, WMMA).
//
// Phase A (parallel):  pack features -> bf16, swizzle Wi/Wh into WMMA-B lane
//                      layout, init out = bd.
// Phase B (parallel):  gi = x @ Wi + bi via v_wmma_f32_16x16x32_bf16, stored
//                      in WMMA C-layout (bf16) for direct accumulator seeding.
// Phase C (2 persistent WGs, one per direction): 2048-step scan.
//   - Wh r/z gates (256KB bf16, swizzled) resident in LDS (320KB/WGP).
//   - Wh n gate (128KB) streamed per step from L2 (fully L2-resident).
//   - h ping-pongs through LDS in WMMA A-layout; 1 barrier/step.
//   - 768 bf16 WMMAs / step / WG, f32 accumulation; exp-based sigmoid/tanh
//     (single v_exp_f32 transcendental each) on the serial critical path.
//   - output projection fused: shfl_xor wave32 reduction + global f32 atomics
//     into out[b,t] (pre-seeded with bd).
//
// Workspace requirement: ~211 MB (gi tensor dominates: 2*2048*32*768 bf16).
// ---------------------------------------------------------------------------

typedef __attribute__((ext_vector_type(16))) __bf16 v16bf;
typedef __attribute__((ext_vector_type(8)))  __bf16 v8bf;
typedef __attribute__((ext_vector_type(8)))  float  v8f;

#define T_LEN 2048
#define BATCH 32
#define HID   256
#define FEAT  64
#define N3H   768
#define NT    48        // N tiles (768/16)
#define KT_H  8         // K tiles over H (256/32)
#define KT_F  2         // K tiles over FEAT (64/32)

// workspace layout (bytes)
#define X_OFF        0ull
#define X_BYTES      ((size_t)T_LEN*BATCH*FEAT*2)
#define WI_DIR_ELEMS ((size_t)NT*KT_F*32*16)
#define WI_OFF       (X_OFF + X_BYTES)
#define WI_BYTES     (2*WI_DIR_ELEMS*2)
#define WH_DIR_ELEMS ((size_t)NT*KT_H*32*16)
#define WH_OFF       (WI_OFF + WI_BYTES)
#define WH_BYTES     (2*WH_DIR_ELEMS*2)
#define GI_OFF       (WH_OFF + WH_BYTES)
#define GI_PER_DIRT  ((size_t)NT*2*32*8)   // elems per (dir,t) = 24576

__device__ __forceinline__ v8f splat8(float x) {
  v8f v;
#pragma unroll
  for (int i = 0; i < 8; ++i) v[i] = x;
  return v;
}

// one transcendental each; correct saturation at +/- infinity
__device__ __forceinline__ float fast_sigmoid(float x) {
  return 1.f / (1.f + __expf(-x));
}
__device__ __forceinline__ float fast_tanh(float x) {
  return 1.f - 2.f / (__expf(2.f * x) + 1.f);
}

// ---------------------------------------------------------------------------
// Swizzle a [K, 768] f32 weight matrix into per-lane WMMA B-operand chunks:
// tiles[ntile][ktile][lane][16 bf16], element e of lane l -> (k = kt*32 +
// 8*(l/16) + e (e<8) or 16 + 8*(l/16) + e-8 (e>=8), n = nt*16 + l%16).
// ---------------------------------------------------------------------------
__global__ void swizzleB(const float* __restrict__ W, __bf16* __restrict__ out,
                         int ktiles) {
  int tid = blockIdx.x * blockDim.x + threadIdx.x;
  int total = NT * ktiles * 32;
  if (tid >= total) return;
  int lane = tid & 31;
  int kt = (tid >> 5) % ktiles;
  int nt = (tid >> 5) / ktiles;
  int hi = lane >> 4;
  int n = nt * 16 + (lane & 15);
  __bf16* dst = out + (size_t)tid * 16;
#pragma unroll
  for (int e = 0; e < 16; ++e) {
    int k = kt * 32 + (e < 8 ? hi * 8 + e : 16 + hi * 8 + (e - 8));
    dst[e] = (__bf16)W[(size_t)k * N3H + n];
  }
}

// ---------------------------------------------------------------------------
// Pack features: X[row = t*32+b][0:64] = { duration[b,t], embed[sid[b,t], :] }
// ---------------------------------------------------------------------------
__global__ void packX(const float* __restrict__ dur, const int* __restrict__ sid,
                      const float* __restrict__ emb, __bf16* __restrict__ X) {
  int row = blockIdx.x * blockDim.x + threadIdx.x;
  if (row >= T_LEN * BATCH) return;
  int b = row & 31, t = row >> 5;
  __bf16* d = X + (size_t)row * FEAT;
  d[0] = (__bf16)dur[(size_t)b * T_LEN + t];
  int s = sid[(size_t)b * T_LEN + t];
  const float* e = emb + (size_t)s * 63;
#pragma unroll 7
  for (int k = 0; k < 63; ++k) d[1 + k] = (__bf16)e[k];
}

__global__ void initOut(float* __restrict__ out, const float* __restrict__ bd) {
  int i = blockIdx.x * blockDim.x + threadIdx.x;
  if (i < T_LEN * BATCH) out[i] = bd[0];
}

// ---------------------------------------------------------------------------
// gi = x @ Wi + bi  (both directions).  grid (T/4, 2), block 256 (8 waves).
// Each WG: 128 rows (4 t-values, 8 M-tiles).  Wave w owns 6 N-tiles, holds
// Wi B-tiles in registers, chains K=2 WMMAs per accumulator.
// Output in WMMA C-layout bf16: GI[dir][t][ntile][mtile][lane][8].
// ---------------------------------------------------------------------------
__global__ __launch_bounds__(256) void giGemm(
    const __bf16* __restrict__ X, const __bf16* __restrict__ WIsw,
    const float* __restrict__ bi_f, const float* __restrict__ bi_b,
    __bf16* __restrict__ GI) {
  int dir = blockIdx.y;
  const __bf16* Wi = WIsw + (size_t)dir * WI_DIR_ELEMS;
  const float* bi = dir ? bi_b : bi_f;
  int w = threadIdx.x >> 5, lane = threadIdx.x & 31;
  int hi = lane >> 4, lo = lane & 15;
  int nbase = w * 6;
  int t0 = blockIdx.x * 4;

  v16bf Bt[6][2];
#pragma unroll
  for (int nt = 0; nt < 6; ++nt)
#pragma unroll
    for (int kt = 0; kt < 2; ++kt)
      Bt[nt][kt] = *(const v16bf*)(Wi + ((((size_t)(nbase + nt)) * KT_F + kt) * 32 + lane) * 16);

  float bival[6];
#pragma unroll
  for (int nt = 0; nt < 6; ++nt) bival[nt] = bi[(nbase + nt) * 16 + lo];

  for (int mt = 0; mt < 8; ++mt) {
    int t = t0 + (mt >> 1);
    int mrow = t0 * 32 + mt * 16;
    // A tile: lane hi-half holds K groups [8*hi, 8*hi+8) and [16+8*hi, 24+8*hi)
    union { v16bf v; v8bf h[2]; } A[2];
    const __bf16* xr = X + ((size_t)mrow + lo) * FEAT;
#pragma unroll
    for (int kt = 0; kt < 2; ++kt) {
      A[kt].h[0] = *(const v8bf*)(xr + kt * 32 + 8 * hi);
      A[kt].h[1] = *(const v8bf*)(xr + kt * 32 + 16 + 8 * hi);
    }
    v8f acc[6];
#pragma unroll
    for (int nt = 0; nt < 6; ++nt) acc[nt] = splat8(bival[nt]);
#pragma unroll
    for (int kt = 0; kt < 2; ++kt)
#pragma unroll
      for (int nt = 0; nt < 6; ++nt)
        acc[nt] = __builtin_amdgcn_wmma_f32_16x16x32_bf16(
            false, A[kt].v, false, Bt[nt][kt], (short)0, acc[nt], false, false);
#pragma unroll
    for (int nt = 0; nt < 6; ++nt) {
      v8bf o;
#pragma unroll
      for (int e = 0; e < 8; ++e) o[e] = (__bf16)acc[nt][e];
      size_t idx = ((((size_t)dir * T_LEN + t) * NT + (nbase + nt)) * 2 + (mt & 1)) * 32 + lane;
      *(v8bf*)(GI + idx * 8) = o;
    }
  }
}

// ---------------------------------------------------------------------------
// Recurrent scan.  grid 2 (direction), block 256 (8 waves), persistent.
// Wave w owns hidden columns j in [32w, 32w+32): r-tiles {2w,2w+1},
// z-tiles {16+2w,..}, n-tiles {32+2w,..}.  Per step per wave: 96 WMMAs.
// ---------------------------------------------------------------------------
extern __shared__ char smem[];

__global__ __launch_bounds__(256) void scanGRU(
    const __bf16* __restrict__ WHsw, const __bf16* __restrict__ GI,
    const float* __restrict__ bhn_f, const float* __restrict__ bhn_b,
    const float* __restrict__ Wd, float* __restrict__ out) {
  int dir = blockIdx.x;
  const __bf16* Wh = WHsw + (size_t)dir * WH_DIR_ELEMS;
  const float* bhn = dir ? bhn_b : bhn_f;
  int w = threadIdx.x >> 5, lane = threadIdx.x & 31;
  int hi = lane >> 4, lo = lane & 15;

  __bf16* whrz = (__bf16*)smem;                           // 32*8*32*16 bf16 = 256KB
  __bf16* hA = (__bf16*)(smem + 32 * KT_H * 32 * 16 * 2); // 2 * 8K bf16 = 32KB
  const int HA_ELEMS = KT_H * 2 * 32 * 16;                // 8192

  // one-time: copy swizzled r/z weights (ntiles 0..31) into LDS
  for (int i = threadIdx.x; i < 32 * KT_H * 32; i += 256)
    *(v16bf*)(whrz + (size_t)i * 16) = *(const v16bf*)(Wh + (size_t)i * 16);
  // zero h buffer 0 (h0 = 0)
  for (int i = threadIdx.x; i < HA_ELEMS / 2; i += 256) ((unsigned int*)hA)[i] = 0u;

  float wdv[2], bhnv[2];
  int stBase[2];
#pragma unroll
  for (int tn = 0; tn < 2; ++tn) {
    int j = 32 * w + 16 * tn + lo;
    wdv[tn] = Wd[dir * HID + j];
    bhnv[tn] = bhn[j];
    // h-store swizzle into A-layout: ktile = w, kl = 16*tn+lo
    int kl = 16 * tn + lo;
    int hip = (kl >> 3) & 1;
    int ep = (kl & 7) + ((kl >> 4) & 1) * 8;
    stBase[tn] = ((w * 2) * 32 + 8 * hi + 16 * hip) * 16 + ep;
  }

  v8f h_c[2][2];
#pragma unroll
  for (int tn = 0; tn < 2; ++tn)
#pragma unroll
    for (int m = 0; m < 2; ++m) h_c[tn][m] = splat8(0.f);

  __syncthreads();

  int cur = 0;
  for (int s = 0; s < T_LEN; ++s) {
    int t = dir ? (T_LEN - 1 - s) : s;
    const __bf16* gi_t = GI + ((size_t)dir * T_LEN + t) * GI_PER_DIRT;
    // prefetch next step's gi slices (r/z/n regions) -> global_prefetch_b8
    if (s + 1 < T_LEN) {
      int t2 = dir ? (T_LEN - 2 - s) : (s + 1);
      const __bf16* g2 = GI + ((size_t)dir * T_LEN + t2) * GI_PER_DIRT;
      __builtin_prefetch(g2 + ((size_t)(2 * w) * 2 * 256) + lane * 8, 0, 2);
      __builtin_prefetch(g2 + ((size_t)(16 + 2 * w) * 2 * 256) + lane * 8, 0, 2);
      __builtin_prefetch(g2 + ((size_t)(32 + 2 * w) * 2 * 256) + lane * 8, 0, 2);
    }
    __bf16* hAc = hA + cur * HA_ELEMS;
    __bf16* hAn = hA + (1 - cur) * HA_ELEMS;

    // load h(t-1) as A tiles from LDS
    v16bf A[2][KT_H];
#pragma unroll
    for (int m = 0; m < 2; ++m)
#pragma unroll
      for (int k = 0; k < KT_H; ++k)
        A[m][k] = *(v16bf*)(hAc + ((k * 2 + m) * 32 + lane) * 16);

    float proj[2][8];
#pragma unroll
    for (int m = 0; m < 2; ++m)
#pragma unroll
      for (int e = 0; e < 8; ++e) proj[m][e] = 0.f;

#pragma unroll
    for (int tn = 0; tn < 2; ++tn) {
      int ntr = 2 * w + tn, ntz = 16 + 2 * w + tn, ntn = 32 + 2 * w + tn;
      v8f ar[2], az[2], an[2], inn[2];
#pragma unroll
      for (int m = 0; m < 2; ++m) {
        v8bf gr = *(const v8bf*)(gi_t + ((size_t)(ntr * 2 + m) * 32 + lane) * 8);
        v8bf gz = *(const v8bf*)(gi_t + ((size_t)(ntz * 2 + m) * 32 + lane) * 8);
        v8bf gn = *(const v8bf*)(gi_t + ((size_t)(ntn * 2 + m) * 32 + lane) * 8);
#pragma unroll
        for (int e = 0; e < 8; ++e) {
          ar[m][e] = (float)gr[e];
          az[m][e] = (float)gz[e];
          inn[m][e] = (float)gn[e];
        }
        an[m] = splat8(bhnv[tn]);
      }
#pragma unroll
      for (int k = 0; k < KT_H; ++k) {
        v16bf br = *(v16bf*)(whrz + (((size_t)ntr * KT_H + k) * 32 + lane) * 16);
        v16bf bz = *(v16bf*)(whrz + (((size_t)ntz * KT_H + k) * 32 + lane) * 16);
        v16bf bn = *(const v16bf*)(Wh + (((size_t)ntn * KT_H + k) * 32 + lane) * 16);
#pragma unroll
        for (int m = 0; m < 2; ++m) {
          ar[m] = __builtin_amdgcn_wmma_f32_16x16x32_bf16(false, A[m][k], false, br,
                                                          (short)0, ar[m], false, false);
          az[m] = __builtin_amdgcn_wmma_f32_16x16x32_bf16(false, A[m][k], false, bz,
                                                          (short)0, az[m], false, false);
          an[m] = __builtin_amdgcn_wmma_f32_16x16x32_bf16(false, A[m][k], false, bn,
                                                          (short)0, an[m], false, false);
        }
      }
#pragma unroll
      for (int m = 0; m < 2; ++m) {
#pragma unroll
        for (int e = 0; e < 8; ++e) {
          float r = fast_sigmoid(ar[m][e]);
          float z = fast_sigmoid(az[m][e]);
          float n = fast_tanh(inn[m][e] + r * an[m][e]);
          float hn = (1.f - z) * n + z * h_c[tn][m][e];
          h_c[tn][m][e] = hn;
          proj[m][e] += hn * wdv[tn];
          hAn[stBase[tn] + m * 512 + e * 16] = (__bf16)hn;
        }
      }
    }

    // fused projection: reduce over this wave's 32 j-columns, atomically add
#pragma unroll
    for (int m = 0; m < 2; ++m)
#pragma unroll
      for (int e = 0; e < 8; ++e) {
        float v = proj[m][e];
        v += __shfl_xor(v, 1, 32);
        v += __shfl_xor(v, 2, 32);
        v += __shfl_xor(v, 4, 32);
        v += __shfl_xor(v, 8, 32);
        if (lo == 0) {
          int b = m * 16 + e + 8 * hi;
          atomicAdd(out + (size_t)b * T_LEN + t, v);
        }
      }
    __syncthreads();
    cur ^= 1;
  }
}

// ---------------------------------------------------------------------------
extern "C" void kernel_launch(void* const* d_in, const int* in_sizes, int n_in,
                              void* d_out, int out_size, void* d_ws, size_t ws_size,
                              hipStream_t stream) {
  const float* dur   = (const float*)d_in[0];
  const int*   sid   = (const int*)d_in[1];
  const float* emb   = (const float*)d_in[2];
  const float* Wi_f  = (const float*)d_in[3];
  const float* Wh_f  = (const float*)d_in[4];
  const float* bi_f  = (const float*)d_in[5];
  const float* bhn_f = (const float*)d_in[6];
  const float* Wi_b  = (const float*)d_in[7];
  const float* Wh_b  = (const float*)d_in[8];
  const float* bi_b  = (const float*)d_in[9];
  const float* bhn_b = (const float*)d_in[10];
  const float* Wd    = (const float*)d_in[11];
  const float* bd    = (const float*)d_in[12];
  float* out = (float*)d_out;

  char* ws = (char*)d_ws;
  __bf16* X    = (__bf16*)(ws + X_OFF);
  __bf16* WIsw = (__bf16*)(ws + WI_OFF);
  __bf16* WHsw = (__bf16*)(ws + WH_OFF);
  __bf16* GI   = (__bf16*)(ws + GI_OFF);

  // Phase A: swizzle weights, pack features, seed output with bd
  swizzleB<<<(NT * KT_F * 32 + 255) / 256, 256, 0, stream>>>(Wi_f, WIsw, KT_F);
  swizzleB<<<(NT * KT_F * 32 + 255) / 256, 256, 0, stream>>>(Wi_b, WIsw + WI_DIR_ELEMS, KT_F);
  swizzleB<<<(NT * KT_H * 32 + 255) / 256, 256, 0, stream>>>(Wh_f, WHsw, KT_H);
  swizzleB<<<(NT * KT_H * 32 + 255) / 256, 256, 0, stream>>>(Wh_b, WHsw + WH_DIR_ELEMS, KT_H);
  packX<<<(T_LEN * BATCH + 255) / 256, 256, 0, stream>>>(dur, sid, emb, X);
  initOut<<<(T_LEN * BATCH + 255) / 256, 256, 0, stream>>>(out, bd);

  // Phase B: input-gate GEMM (both directions)
  dim3 gGrid(T_LEN / 4, 2);
  giGemm<<<gGrid, 256, 0, stream>>>(X, WIsw, bi_f, bi_b, GI);

  // Phase C: persistent recurrent scan, one WG per direction
  const int smemBytes = 32 * KT_H * 32 * 16 * 2 + 2 * (KT_H * 2 * 32 * 16) * 2; // 294912
  hipFuncSetAttribute(reinterpret_cast<const void*>(scanGRU),
                      hipFuncAttributeMaxDynamicSharedMemorySize, smemBytes);
  scanGRU<<<2, 256, smemBytes, stream>>>(WHsw, GI, bhn_f, bhn_b, Wd, out);
}
